// GlobalMHRA_3118146257165
// MI455X (gfx1250) — matrix operand order repaired and checked by reading, not compiled, not run on hardware
//
#include <hip/hip_runtime.h>

typedef __attribute__((ext_vector_type(16))) _Float16 v16h;
typedef __attribute__((ext_vector_type(8)))  _Float16 v8h;
typedef __attribute__((ext_vector_type(8)))  float    v8f;

namespace {
constexpr int   kB     = 2;
constexpr int   kC     = 128;   // channels (LN + GEMM K-dim)
constexpr int   kN     = 2560;  // T*H*W positions
constexpr int   kHeads = 4;
constexpr int   kD     = 64;    // head dim
constexpr int   kInner = 256;   // heads * d
constexpr int   kQKVR  = 768;   // 3 * inner
constexpr float kEps   = 1e-5f;
constexpr float kScale = 0.125f; // 64^-0.5
}

__device__ __forceinline__ v8f zero_v8f() {
  v8f r;
#pragma unroll
  for (int i = 0; i < 8; ++i) r[i] = 0.0f;
  return r;
}

// A fragment (16x32 f16, MxK) from a row-major matrix (stride ld, K contiguous).
// Lane m (0..15) holds row m; lane-half selects K subgroups per ISA 7.12.2.
__device__ __forceinline__ v16h frag_a(const _Float16* base, int ld, int row0, int k0, int lane) {
  const int m = lane & 15, hi = lane >> 4;
  const _Float16* p = base + (size_t)(row0 + m) * ld + k0;
  v8h lo = *(const v8h*)(p + 8 * hi);
  v8h hh = *(const v8h*)(p + 16 + 8 * hi);
  v16h r;
#pragma unroll
  for (int i = 0; i < 8; ++i) { r[i] = lo[i]; r[8 + i] = hh[i]; }
  return r;
}

// B fragment (32x16 f16, KxN) from an N-major matrix ("B^T": row = n, K contiguous).
// Lane n holds column n; lane-half selects K subgroups (8h..8h+7, 16+8h..16+8h+7).
__device__ __forceinline__ v16h frag_b(const _Float16* bt, int ld, int col0, int k0, int lane) {
  const int n = lane & 15, hi = lane >> 4;
  const _Float16* p = bt + (size_t)(col0 + n) * ld + k0 + 8 * hi;
  v8h x0 = *(const v8h*)(p);
  v8h x1 = *(const v8h*)(p + 16);
  v16h r;
#pragma unroll
  for (int i = 0; i < 8; ++i) { r[i] = x0[i]; r[8 + i] = x1[i]; }
  return r;
}

__device__ __forceinline__ v8f wmma16(v16h a, v16h b, v8f c) {
  return __builtin_amdgcn_wmma_f32_16x16x32_f16(false, a, false, b, (short)0, c, false, false);
}

// ---------------------------------------------------------------------------
// 1) f32 -> f16 weight conversion
// ---------------------------------------------------------------------------
__global__ void __launch_bounds__(256) cvt_f16_kernel(const float* __restrict__ s,
                                                      _Float16* __restrict__ d, int n) {
  int i = blockIdx.x * 256 + threadIdx.x;
  if (i < n) d[i] = (_Float16)s[i];
}

// ---------------------------------------------------------------------------
// 2) Channel LayerNorm: x[B][C][n] -> xnT[B][n][C] (f16, C contiguous for GEMM-B)
// ---------------------------------------------------------------------------
__global__ void __launch_bounds__(256) ln_kernel(const float* __restrict__ x,
                                                 const float* __restrict__ g,
                                                 const float* __restrict__ bch,
                                                 _Float16* __restrict__ xnT) {
  int p = blockIdx.x * 256 + threadIdx.x;       // 0 .. B*n-1
  if (p >= kB * kN) return;
  int b = p / kN, pos = p % kN;
  const float* xb = x + (size_t)b * kC * kN + pos;
  float s1 = 0.f, s2 = 0.f;
#pragma unroll 4
  for (int c = 0; c < kC; ++c) {
    float v = xb[(size_t)c * kN];
    s1 += v; s2 += v * v;
  }
  float mean = s1 * (1.0f / kC);
  float var  = s2 * (1.0f / kC) - mean * mean;
  var = var < 0.f ? 0.f : var;
  float inv = 1.0f / (sqrtf(var) + kEps);
  _Float16* o = xnT + (size_t)p * kC;
#pragma unroll 4
  for (int c = 0; c < kC; ++c) {
    float v = (xb[(size_t)c * kN] - mean) * inv * g[c] + bch[c];
    o[c] = (_Float16)v;
  }
}

// ---------------------------------------------------------------------------
// 3) QKV projection: W(768x128) @ xnT^T(128x2560) per batch; route to
//    q[b][h][pos][d] (scaled), k[b][h][pos][d], vT[b][h][d][pos].
//    One wave per 16(M) x 64(N) tile.
// ---------------------------------------------------------------------------
__global__ void __launch_bounds__(32) qkv_kernel(const _Float16* __restrict__ w,
                                                 const _Float16* __restrict__ xnT,
                                                 _Float16* __restrict__ q16,
                                                 _Float16* __restrict__ k16,
                                                 _Float16* __restrict__ vT16) {
  const int lane = threadIdx.x;
  const int ng = blockIdx.x;   // 0..39 (groups of 4 n-tiles)
  const int mt = blockIdx.y;   // 0..47
  const int b  = blockIdx.z;
  const _Float16* bt = xnT + (size_t)b * kN * kC;

  v8f acc[4];
#pragma unroll
  for (int gI = 0; gI < 4; ++gI) acc[gI] = zero_v8f();

#pragma unroll
  for (int kc = 0; kc < kC / 32; ++kc) {
    v16h a = frag_a(w, kC, mt * 16, kc * 32, lane);
#pragma unroll
    for (int gI = 0; gI < 4; ++gI) {
      v16h bf = frag_b(bt, kC, (ng * 4 + gI) * 16, kc * 32, lane);
      acc[gI] = wmma16(a, bf, acc[gI]);
    }
  }

  const int nn = lane & 15, hi = lane >> 4;
  const int o0  = mt * 16;
  const int sec = o0 >> 8;          // 0=q, 1=k, 2=v (uniform per block)
  const int io0 = o0 & 255;
  const int h   = io0 >> 6;
  const int d0  = io0 & 63;
  const size_t bh = (size_t)b * kHeads + h;

#pragma unroll
  for (int gI = 0; gI < 4; ++gI) {
    int pos = (ng * 4 + gI) * 16 + nn;
#pragma unroll
    for (int j = 0; j < 8; ++j) {
      float v = acc[gI][j];
      int d = d0 + 8 * hi + j;
      if (sec == 0)
        q16[(bh * kN + pos) * kD + d] = (_Float16)(v * kScale);
      else if (sec == 1)
        k16[(bh * kN + pos) * kD + d] = (_Float16)v;
      else
        vT16[(bh * kD + d) * kN + pos] = (_Float16)v;
    }
  }
}

// ---------------------------------------------------------------------------
// 4) Fused attention. One workgroup (8 waves) per (b, h, 16-query block).
//    LDS: S = 16 x 2560 f32 scores (160KB) + 8 x 16 x 64 f32 O-partials (32KB).
//    S never touches HBM.
// ---------------------------------------------------------------------------
// P@V fragment built on the fly: P = exp(S - rowmax) / rowsum, f16.
__device__ __forceinline__ v16h frag_a_p(const float* S, int j0, int lane,
                                         float mrow, float rinv) {
  const int hi = lane >> 4;
  const float* p = S + (size_t)(lane & 15) * kN + j0 + 8 * hi;
  v16h r;
#pragma unroll
  for (int i = 0; i < 8; ++i) r[i] = (_Float16)(__expf(p[i] - mrow) * rinv);
#pragma unroll
  for (int i = 0; i < 8; ++i) r[8 + i] = (_Float16)(__expf(p[16 + i] - mrow) * rinv);
  return r;
}

__global__ void __launch_bounds__(256) attn_kernel(const _Float16* __restrict__ q16,
                                                   const _Float16* __restrict__ k16,
                                                   const _Float16* __restrict__ vT16,
                                                   _Float16* __restrict__ att16) {
  const int qt = blockIdx.x;            // 0..159
  const int h  = blockIdx.y;            // 0..3
  const int b  = blockIdx.z;            // 0..1
  const int tid = threadIdx.x;
  const int wv = tid >> 5, lane = tid & 31;
  const int nn = lane & 15, hi = lane >> 4;

  extern __shared__ float4 dynsm[];
  float* S     = (float*)dynsm;          // 16 * 2560 floats
  float* Opart = S + 16 * kN;            // 8 * 16 * 64 floats
  __shared__ float red[16 * 16];
  __shared__ float rowmax[16];
  __shared__ float rowsum[16];

  const size_t bh = (size_t)b * kHeads + h;
  const _Float16* Q  = q16 + bh * kN * kD;
  const _Float16* K  = k16 + bh * kN * kD;
  const _Float16* VT = vT16 + bh * kD * kN;
  const int q0 = qt * 16;

  // --- S = Q K^T : wave wv handles keys [wv*320, wv*320+320) -----------------
  v16h aq0 = frag_a(Q, kD, q0, 0, lane);
  v16h aq1 = frag_a(Q, kD, q0, 32, lane);
  for (int t = 0; t < 20; ++t) {
    int j0 = (wv * 20 + t) * 16;
    if (t + 1 < 20) __builtin_prefetch(K + (size_t)(j0 + 16) * kD, 0, 0);
    v16h b0 = frag_b(K, kD, j0, 0, lane);
    v16h b1 = frag_b(K, kD, j0, 32, lane);
    v8f c = zero_v8f();
    c = wmma16(aq0, b0, c);
    c = wmma16(aq1, b1, c);
#pragma unroll
    for (int j = 0; j < 8; ++j) S[(size_t)(8 * hi + j) * kN + j0 + nn] = c[j];
  }
  __syncthreads();

  // --- softmax statistics: 16 threads per row --------------------------------
  const int r = tid >> 4, cc = tid & 15;
  float* Srow = S + (size_t)r * kN;
  float mx = -3.0e38f;
  for (int j = cc; j < kN; j += 16) mx = fmaxf(mx, Srow[j]);
  red[r * 16 + cc] = mx;
  __syncthreads();
  if (cc == 0) {
    float m = red[r * 16];
#pragma unroll
    for (int i = 1; i < 16; ++i) m = fmaxf(m, red[r * 16 + i]);
    rowmax[r] = m;
  }
  __syncthreads();
  float m = rowmax[r];
  float sum = 0.f;
  for (int j = cc; j < kN; j += 16) sum += __expf(Srow[j] - m);
  red[r * 16 + cc] = sum;
  __syncthreads();
  if (cc == 0) {
    float s = 0.f;
#pragma unroll
    for (int i = 0; i < 16; ++i) s += red[r * 16 + i];
    rowsum[r] = s;
  }
  __syncthreads();

  // --- O = P @ V : wave wv handles j-range [wv*320, wv*320+320) --------------
  const float mrow = rowmax[lane & 15];
  const float rinv = 1.0f / rowsum[lane & 15];
  v8f oc[4];
#pragma unroll
  for (int dt = 0; dt < 4; ++dt) oc[dt] = zero_v8f();
  for (int t = 0; t < 10; ++t) {
    int j0 = wv * 320 + t * 32;
    v16h ap = frag_a_p(S, j0, lane, mrow, rinv);
#pragma unroll
    for (int dt = 0; dt < 4; ++dt) {
      v16h bv = frag_b(VT, kN, dt * 16, j0, lane);
      oc[dt] = wmma16(ap, bv, oc[dt]);
    }
  }
#pragma unroll
  for (int dt = 0; dt < 4; ++dt)
#pragma unroll
    for (int j = 0; j < 8; ++j)
      Opart[(size_t)(wv * 16 + 8 * hi + j) * kD + dt * 16 + nn] = oc[dt][j];
  __syncthreads();

  // --- reduce 8 wave-partials, emit att[b][pos][h*64+d] (f16, inner-major) ---
  for (int e = tid; e < 16 * kD; e += 256) {
    int mm = e >> 6, d = e & 63;
    float s = 0.f;
#pragma unroll
    for (int w = 0; w < 8; ++w) s += Opart[(size_t)(w * 16 + mm) * kD + d];
    att16[((size_t)b * kN + q0 + mm) * kInner + h * kD + d] = (_Float16)s;
  }
}

// ---------------------------------------------------------------------------
// 5) Output projection: Wout(128x256) @ att^T(256x2560) + bias -> out[b][C][n]
// ---------------------------------------------------------------------------
__global__ void __launch_bounds__(32) outproj_kernel(const _Float16* __restrict__ wo,
                                                     const _Float16* __restrict__ att16,
                                                     const float* __restrict__ bout,
                                                     float* __restrict__ out) {
  const int lane = threadIdx.x;
  const int ng = blockIdx.x;   // 0..39
  const int mt = blockIdx.y;   // 0..7
  const int b  = blockIdx.z;
  const _Float16* bt = att16 + (size_t)b * kN * kInner;

  v8f acc[4];
#pragma unroll
  for (int gI = 0; gI < 4; ++gI) acc[gI] = zero_v8f();

#pragma unroll
  for (int kc = 0; kc < kInner / 32; ++kc) {
    v16h a = frag_a(wo, kInner, mt * 16, kc * 32, lane);
#pragma unroll
    for (int gI = 0; gI < 4; ++gI) {
      v16h bf = frag_b(bt, kInner, (ng * 4 + gI) * 16, kc * 32, lane);
      acc[gI] = wmma16(a, bf, acc[gI]);
    }
  }

  const int nn = lane & 15, hi = lane >> 4;
#pragma unroll
  for (int gI = 0; gI < 4; ++gI) {
    int pos = (ng * 4 + gI) * 16 + nn;
#pragma unroll
    for (int j = 0; j < 8; ++j) {
      int o = mt * 16 + 8 * hi + j;
      out[((size_t)b * kC + o) * kN + pos] = acc[gI][j] + bout[o];
    }
  }
}

// ---------------------------------------------------------------------------
extern "C" void kernel_launch(void* const* d_in, const int* in_sizes, int n_in,
                              void* d_out, int out_size, void* d_ws, size_t ws_size,
                              hipStream_t stream) {
  (void)in_sizes; (void)n_in; (void)out_size; (void)ws_size;
  const float* x    = (const float*)d_in[0];
  const float* g    = (const float*)d_in[1];
  const float* bch  = (const float*)d_in[2];
  const float* wqkv = (const float*)d_in[3];
  const float* wout = (const float*)d_in[4];
  const float* bout = (const float*)d_in[5];
  float* out = (float*)d_out;

  char* ws = (char*)d_ws;
  size_t off = 0;
  auto carve = [&](size_t bytes) -> char* {
    char* p = ws + off;
    off += (bytes + 255) & ~(size_t)255;
    return p;
  };
  _Float16* xnT  = (_Float16*)carve((size_t)kB * kN * kC * 2);
  _Float16* wq16 = (_Float16*)carve((size_t)kQKVR * kC * 2);
  _Float16* wo16 = (_Float16*)carve((size_t)kC * kInner * 2);
  _Float16* q16  = (_Float16*)carve((size_t)kB * kHeads * kN * kD * 2);
  _Float16* k16  = (_Float16*)carve((size_t)kB * kHeads * kN * kD * 2);
  _Float16* vT16 = (_Float16*)carve((size_t)kB * kHeads * kD * kN * 2);
  _Float16* att16= (_Float16*)carve((size_t)kB * kN * kInner * 2);

  constexpr int kAttnShmem = 16 * kN * 4 + 8 * 16 * kD * 4;  // 196608 bytes
  static_assert(kAttnShmem <= 320 * 1024, "LDS budget");
  (void)hipFuncSetAttribute((const void*)attn_kernel,
                            hipFuncAttributeMaxDynamicSharedMemorySize, kAttnShmem);

  cvt_f16_kernel<<<(kQKVR * kC + 255) / 256, 256, 0, stream>>>(wqkv, wq16, kQKVR * kC);
  cvt_f16_kernel<<<(kC * kInner + 255) / 256, 256, 0, stream>>>(wout, wo16, kC * kInner);
  ln_kernel<<<(kB * kN + 255) / 256, 256, 0, stream>>>(x, g, bch, xnT);
  qkv_kernel<<<dim3(kN / 64, kQKVR / 16, kB), 32, 0, stream>>>(wq16, xnT, q16, k16, vT16);
  attn_kernel<<<dim3(kN / 16, kHeads, kB), 256, kAttnShmem, stream>>>(q16, k16, vT16, att16);
  outproj_kernel<<<dim3(kN / 64, kC / 16, kB), 32, 0, stream>>>(wo16, att16, bout, out);
}
